// ExpertModule_3loss_85495618994905
// MI455X (gfx1250) — matrix operand order repaired and checked by reading, not compiled, not run on hardware
//
#include <hip/hip_runtime.h>
#include <hip/hip_bf16.h>

// ---------------- problem constants (match reference) ----------------
constexpr int Bn   = 4;
constexpr int Tn   = 1024;
constexpr int Dd   = 1024;        // input dim
constexpr int Hh   = 4096;        // hidden dim
constexpr int Ee   = 8;           // experts
constexpr int Ntok = Bn * Tn;     // 4096 tokens
constexpr int HC   = 512;         // hidden chunk per fused pass
constexpr int MT   = 32;          // tokens per FFN block (2 WMMA M-tiles)

// ---------------- WMMA plumbing (gfx1250, wave32) ----------------
typedef __attribute__((ext_vector_type(16))) __bf16 v16bf;
typedef __attribute__((ext_vector_type(8)))  float  v8f;

union BF16x16 { unsigned u[8]; uint4 q[2]; v16bf v; };

__device__ __forceinline__ v8f vzero() {
  v8f z;
#pragma unroll
  for (int i = 0; i < 8; ++i) z[i] = 0.f;
  return z;
}

__device__ __forceinline__ unsigned bfbits(float f) {
  unsigned u = __float_as_uint(f);
  return (u + 0x7fffu + ((u >> 16) & 1u)) >> 16;   // RNE-ish truncate
}
__device__ __forceinline__ unsigned packbf(float lo, float hi) {
  return bfbits(lo) | (bfbits(hi) << 16);
}

__device__ __forceinline__ v8f wmma_bf16(v16bf a, v16bf b, v8f c) {
  // (neg_a, A, neg_b, B, c_mod, C, reuse_a, reuse_b)
  return __builtin_amdgcn_wmma_f32_16x16x32_bf16(false, a, false, b, (short)0, c,
                                                 false, false);
}

// Load 8 contiguous packed-bf16 uints (32 B) -> one WMMA operand
// (LDS: 2x ds_load_b128, global: 2x global_load_b128).
__device__ __forceinline__ v16bf ld8u(const unsigned* s) {
  BF16x16 r;
  const uint4* p = (const uint4*)s;
  r.q[0] = p[0];
  r.q[1] = p[1];
  return r.v;
}

// A-operand from a packed-bf16 row-major row (Gram kernel).
__device__ __forceinline__ v16bf ldA_row(const unsigned* rowu, int k0, int kh) {
  BF16x16 r;
  const uint4* p = (const uint4*)(rowu + (k0 >> 1) + 8 * kh);
  r.q[0] = p[0];      // uints b0 .. b0+3
  r.q[1] = p[1];      // uints b0+4 .. b0+7
  return r.v;
}
// B-operand where column n = contiguous row (B layout: K pairs {k0+16kh+2j}).
__device__ __forceinline__ v16bf ldB_row(const unsigned* rowu, int k0, int kh) {
  return ld8u(rowu + (k0 >> 1) + 8 * kh);
}

// ---------------- kernel 0: zero accumulators ----------------
__global__ void moe_init_kernel(int* __restrict__ counts, float* __restrict__ fbufs) {
  int t = threadIdx.x;
  if (t < Ee) counts[t] = 0;
  if (t < 4 * Ee) fbufs[t] = 0.f;   // sum_p, psum, p2sum, off_acc
}

// ---------------- kernel 1: weight pre-swizzle (fp32 -> bf16, B-operand layout) ----
// uint index = ((e*KT + kt)*NT + nt)*256 + lane*8 + j
// element pair = W[e][k][n], k = kt*32 + 8*(lane>>4) + (j<4 ? 2j : 16+2(j-4)), n = nt*16 + (lane&15).
__global__ __launch_bounds__(256) void moe_wswz_kernel(
    const float* __restrict__ w, unsigned* __restrict__ wsw,
    int N, int ntShift, int ktShift) {
  long idx = (long)blockIdx.x * 256 + threadIdx.x;
  int  j   = (int)(idx & 7);
  int  l   = (int)((idx >> 3) & 31);
  long tile = idx >> 8;
  int  nt  = (int)(tile & ((1 << ntShift) - 1));
  long t2  = tile >> ntShift;
  int  kt  = (int)(t2 & ((1 << ktShift) - 1));
  int  e   = (int)(t2 >> ktShift);
  int  n   = nt * 16 + (l & 15);
  long k   = (long)kt * 32 + 8 * (l >> 4) + (j < 4 ? 2 * j : 16 + 2 * (j - 4));
  const float* we = w + (long)e * ((long)(1 << ktShift) * 32) * N;
  wsw[idx] = packbf(we[k * N + n], we[(k + 1) * N + n]);
}

// ---------------- kernel 2: x pre-swizzle (fp32 -> per-token A-operand fragments) ---
// Per token: 512 uints; index = kt*16 + kh*8 + j holds x[k], x[k+1] where
// k = kt*32 + 8*kh + (j<4 ? 2j : 16+2(j-4)).  Bucket-independent.
__global__ __launch_bounds__(256) void moe_xswz_kernel(
    const float* __restrict__ x, unsigned* __restrict__ xs) {
  int tok = blockIdx.x;
  const float* xr = x + (long)tok * Dd;
  unsigned* o = xs + (long)tok * (Dd / 2);
#pragma unroll
  for (int s = 0; s < 2; ++s) {
    int u  = threadIdx.x + s * 256;
    int kt = u >> 4;
    int kh = (u >> 3) & 1;
    int j  = u & 7;
    int k  = kt * 32 + 8 * kh + (j < 4 ? 2 * j : 16 + 2 * (j - 4));
    o[u] = packbf(xr[k], xr[k + 1]);
  }
}

// ---------------- kernel 3: gating / routing ----------------
__global__ __launch_bounds__(256) void moe_gate_kernel(
    const float* __restrict__ x, const float* __restrict__ gw,
    const float* __restrict__ gb,
    int* __restrict__ counts, float* __restrict__ sum_p,
    float* __restrict__ psum, float* __restrict__ p2sum,
    int* __restrict__ idxbuf) {
  int wv = threadIdx.x >> 5, lane = threadIdx.x & 31;
  int n = blockIdx.x * 8 + wv;
  const float* xr = x + (long)n * Dd;
  float acc[Ee];
#pragma unroll
  for (int e = 0; e < Ee; ++e) acc[e] = 0.f;
  for (int d = lane; d < Dd; d += 32) {
    float xv = xr[d];
    const float* g = gw + d * Ee;
#pragma unroll
    for (int e = 0; e < Ee; ++e) acc[e] = fmaf(xv, g[e], acc[e]);
  }
#pragma unroll
  for (int e = 0; e < Ee; ++e)
#pragma unroll
    for (int off = 16; off; off >>= 1) acc[e] += __shfl_xor(acc[e], off, 32);
  if (lane == 0) {
    float logits[Ee], p[Ee];
    float mx = -1e30f;
#pragma unroll
    for (int e = 0; e < Ee; ++e) { logits[e] = acc[e] + gb[e]; mx = fmaxf(mx, logits[e]); }
    float s = 0.f;
#pragma unroll
    for (int e = 0; e < Ee; ++e) { p[e] = __expf(logits[e] - mx); s += p[e]; }
    float inv = 1.f / s;
    int am = 0;
#pragma unroll
    for (int e = 1; e < Ee; ++e) if (logits[e] > logits[am]) am = e;
#pragma unroll
    for (int e = 0; e < Ee; ++e) {
      float pe = p[e] * inv;
      atomicAdd(&psum[e], pe);
      atomicAdd(&p2sum[e], pe * pe);
    }
    atomicAdd(&sum_p[am], p[am] * inv);
    int slot = atomicAdd(&counts[am], 1);
    idxbuf[am * Ntok + slot] = n;
  }
}

// ---------------- kernel 4: fused sparse expert FFN (WMMA, M=32) ----------------
// grid = (E, Ntok/32); block = 256 (8 waves). Wave w owns output cols [w*128, w*128+128).
// Two 16-row M-tiles share every B operand: 1x B(2xb128) feeds 2 WMMAs.
__global__ __launch_bounds__(256) void moe_ffn_kernel(
    const unsigned* __restrict__ xs,
    const unsigned* __restrict__ w1s, const float* __restrict__ b1,
    const unsigned* __restrict__ w2s, const float* __restrict__ b2,
    const int* __restrict__ counts, const float* __restrict__ sum_p,
    const int* __restrict__ idxbuf,
    float* __restrict__ out, unsigned short* __restrict__ ybf) {
  // relu(h) chunk in A-operand layout: [mt(2)][kt(16)][lane(32)][8]  (32 KB)
  __shared__ __align__(16) unsigned sHs[2 * (HC / 32) * 32 * 8];
  unsigned short* sHs16 = (unsigned short*)sHs;

  int e = blockIdx.x;
  int tile = blockIdx.y;
  int cnt = counts[e];
  if (tile * MT >= cnt) return;
  int rlim = cnt - tile * MT; if (rlim > MT) rlim = MT;
  const int* bucket = idxbuf + e * Ntok + tile * MT;
  float avgp = sum_p[e] / (float)(cnt > 0 ? cnt : 1);

  int tid = threadIdx.x;
  int lane = tid & 31, wv = tid >> 5;
  int m = lane & 15, kh = lane >> 4;

  // Per-lane A-fragment base pointers for the two M-tiles (token-gathered).
  int r0 = m, r1 = 16 + m;
  int tok0 = bucket[r0 < rlim ? r0 : 0];
  int tok1 = bucket[r1 < rlim ? r1 : 0];
  const unsigned* aA = xs + (long)tok0 * (Dd / 2) + kh * 8;
  const unsigned* aB = xs + (long)tok1 * (Dd / 2) + kh * 8;

  v8f yacc[2][8];
#pragma unroll
  for (int mt = 0; mt < 2; ++mt)
#pragma unroll
    for (int t = 0; t < 8; ++t) yacc[mt][t] = vzero();

  const unsigned* w1e = w1s + (long)e * (Dd / 32) * (Hh / 16) * 256;
  const unsigned* w2e = w2s + (long)e * (Hh / 32) * (Dd / 16) * 256;
  const float* b1e = b1 + e * Hh;

  for (int h0 = 0; h0 < Hh; h0 += HC) {
    // ---- GEMM1: h[32 x HC] = X[32 x D] * W1[D x HC-slice]; wave owns 64 cols
    v8f hacc[2][4];
#pragma unroll
    for (int mt = 0; mt < 2; ++mt)
#pragma unroll
      for (int t = 0; t < 4; ++t) hacc[mt][t] = vzero();
    for (int kt = 0; kt < Dd / 32; ++kt) {
      v16bf A0 = ld8u(aA + kt * 16);
      v16bf A1 = ld8u(aB + kt * 16);
      const unsigned* w1t = w1e + (long)kt * (Hh / 16) * 256 + lane * 8;
#pragma unroll
      for (int t = 0; t < 4; ++t) {
        int nt = (h0 + wv * 64 + t * 16) >> 4;
        v16bf B = ld8u(w1t + nt * 256);
        hacc[0][t] = wmma_bf16(A0, B, hacc[0][t]);
        hacc[1][t] = wmma_bf16(A1, B, hacc[1][t]);
      }
    }
    __syncthreads();   // everyone done reading previous sHs before overwrite
    // bias + relu -> LDS in A-operand layout (16-bit scatter stores)
#pragma unroll
    for (int t = 0; t < 4; ++t) {
      int ncol = wv * 64 + t * 16 + m;       // col within the HC chunk
      float bias = b1e[h0 + ncol];
      int kt2  = ncol >> 5;
      int kk   = ncol & 31;
      int kh2  = (kk >> 3) & 1;
      int rem  = kk - 8 * kh2;               // {0..7} or {16..23}
      int j2   = (rem < 8) ? (rem >> 1) : (4 + ((rem - 16) >> 1));
      int half = kk & 1;
#pragma unroll
      for (int mt = 0; mt < 2; ++mt)
#pragma unroll
        for (int r = 0; r < 8; ++r) {
          float hv = hacc[mt][t][r] + bias;
          hv = hv > 0.f ? hv : 0.f;
          int rowc = r + 8 * kh;             // row within this M-tile
          int l2   = rowc + 16 * kh2;
          sHs16[(((mt * (HC / 32) + kt2) * 32 + l2) * 8 + j2) * 2 + half] =
              (unsigned short)bfbits(hv);
        }
    }
    __syncthreads();   // sHs fully written
    // ---- GEMM2: Y[32 x D] += h[32 x HC] * W2[HC-slice x D]; wave owns 128 cols
    for (int kt = 0; kt < HC / 32; ++kt) {
      v16bf A0 = ld8u(sHs + (kt * 32 + lane) * 8);
      v16bf A1 = ld8u(sHs + (((HC / 32) + kt) * 32 + lane) * 8);
      int ktg = (h0 >> 5) + kt;
      const unsigned* w2t = w2e + (long)ktg * (Dd / 16) * 256 + lane * 8;
#pragma unroll
      for (int t = 0; t < 8; ++t) {
        int nt = wv * 8 + t;
        v16bf B = ld8u(w2t + nt * 256);
        yacc[0][t] = wmma_bf16(A0, B, yacc[0][t]);
        yacc[1][t] = wmma_bf16(A1, B, yacc[1][t]);
      }
    }
  }

  // ---- epilogue: +b2; write scaled output (fp32) and packed-bf16 Y for orth loss
#pragma unroll
  for (int t = 0; t < 8; ++t) {
    int c = wv * 128 + t * 16 + m;
    float bias = b2[e * Dd + c];
#pragma unroll
    for (int mt = 0; mt < 2; ++mt)
#pragma unroll
      for (int r = 0; r < 8; ++r) {
        int row = mt * 16 + r + 8 * kh;      // token row within the 32-token tile
        if (row < rlim) {
          int tok = bucket[row];
          float yv = yacc[mt][t][r] + bias;
          out[(long)tok * Dd + c] = avgp * yv;
          ybf[(long)tok * Dd + c] = (unsigned short)bfbits(yv);
        }
      }
  }
}

// ---------------- kernel 5: orthogonality loss via token Gram (WMMA) ----------------
// off_sq_sum[e] = sum_{i != j, routed} (y_i . y_j)^2  (one wave per (e, i-tile))
__global__ __launch_bounds__(32) void moe_orth_kernel(
    const unsigned* __restrict__ ybfu, const int* __restrict__ counts,
    const int* __restrict__ idxbuf, float* __restrict__ off_acc) {
  int e = blockIdx.x, ti = blockIdx.y;
  int cnt = counts[e];
  if (ti * 16 >= cnt) return;
  const int* bucket = idxbuf + e * Ntok;
  int lane = threadIdx.x, m = lane & 15, kh = lane >> 4;

  int ia = ti * 16 + m;
  const unsigned* arow = ybfu + (long)bucket[ia < cnt ? ia : 0] * (Dd / 2);

  int ntj = (cnt + 15) >> 4;
  float local = 0.f;
  for (int tj = 0; tj < ntj; ++tj) {
    int jb = tj * 16 + m;
    const unsigned* brow = ybfu + (long)bucket[jb < cnt ? jb : 0] * (Dd / 2);
    v8f acc = vzero();
    for (int k0 = 0; k0 < Dd; k0 += 32) {
      v16bf A = ldA_row(arow, k0, kh);
      v16bf B = ldB_row(brow, k0, kh);
      acc = wmma_bf16(A, B, acc);
    }
#pragma unroll
    for (int r = 0; r < 8; ++r) {
      int i = ti * 16 + r + 8 * kh;
      int j = tj * 16 + m;
      if (i < cnt && j < cnt && i != j) local += acc[r] * acc[r];
    }
  }
#pragma unroll
  for (int off = 16; off; off >>= 1) local += __shfl_xor(local, off, 32);
  if (lane == 0) atomicAdd(&off_acc[e], local);
}

// ---------------- kernel 6: combine losses ----------------
__global__ void moe_loss_kernel(const int* __restrict__ counts,
                                const float* __restrict__ psum,
                                const float* __restrict__ p2sum,
                                const float* __restrict__ offac,
                                float* __restrict__ loss_out) {
  if (threadIdx.x != 0 || blockIdx.x != 0) return;
  float aux = 0.f, orth = 0.f, var = 0.f;
  for (int e = 0; e < Ee; ++e) {
    float cf = (float)counts[e];
    float u = cf / (float)Ntok;
    aux += u * u;
    if (counts[e] > 0) orth += offac[e] / (cf * cf);
    float mu = psum[e] / (float)Ntok;
    var += p2sum[e] - (float)Ntok * mu * mu;   // sum_n (p - mu)^2 for this column
  }
  aux *= (float)Ee;
  var = -var / (float)(Ntok * Ee);
  *loss_out = 0.001f * aux + 0.001f * orth + 0.001f * var;
}

// ---------------- host-side launch ----------------
extern "C" void kernel_launch(void* const* d_in, const int* in_sizes, int n_in,
                              void* d_out, int out_size, void* d_ws, size_t ws_size,
                              hipStream_t stream) {
  (void)in_sizes; (void)n_in; (void)out_size; (void)ws_size;
  const float* x  = (const float*)d_in[0];
  const float* gw = (const float*)d_in[1];
  const float* gb = (const float*)d_in[2];
  const float* w1 = (const float*)d_in[3];
  const float* b1 = (const float*)d_in[4];
  const float* w2 = (const float*)d_in[5];
  const float* b2 = (const float*)d_in[6];
  float* out = (float*)d_out;
  float* loss_out = out + (long)Ntok * Dd;

  char* ws = (char*)d_ws;
  int*   counts = (int*)ws;                                   // [E]
  float* sum_p  = (float*)(ws + 32);                          // [E]
  float* psum   = (float*)(ws + 64);                          // [E]
  float* p2sum  = (float*)(ws + 96);                          // [E]
  float* offac  = (float*)(ws + 128);                         // [E]
  int*   idxbuf = (int*)(ws + 256);                           // [E * Ntok]
  size_t off_ybf = 256 + (size_t)Ee * Ntok * sizeof(int);     // 131328
  unsigned short* ybf = (unsigned short*)(ws + off_ybf);      // [Ntok * D] bf16 (8 MB)
  size_t off_xs = off_ybf + (size_t)Ntok * Dd * 2;
  unsigned* xs = (unsigned*)(ws + off_xs);                    // [Ntok * D/2] uints (8 MB)
  size_t off_w1s = off_xs + (size_t)Ntok * Dd * 2;
  unsigned* w1s = (unsigned*)(ws + off_w1s);                  // E*D*H/2 uints (64 MB)
  size_t off_w2s = off_w1s + (size_t)Ee * Dd * Hh * 2;
  unsigned* w2s = (unsigned*)(ws + off_w2s);                  // E*H*D/2 uints (64 MB)

  moe_init_kernel<<<1, 64, 0, stream>>>(counts, sum_p);

  // Pre-swizzle weights into WMMA B-operand bf16 layout.
  long n1 = (long)Ee * Dd * Hh / 2;   // w1: K=D (ktShift 5), N=H (ntShift 8)
  moe_wswz_kernel<<<(unsigned)(n1 / 256), 256, 0, stream>>>(w1, w1s, Hh, 8, 5);
  long n2 = (long)Ee * Hh * Dd / 2;   // w2: K=H (ktShift 7), N=D (ntShift 6)
  moe_wswz_kernel<<<(unsigned)(n2 / 256), 256, 0, stream>>>(w2, w2s, Dd, 6, 7);

  // Pre-swizzle x into per-token A-operand fragments.
  moe_xswz_kernel<<<Ntok, 256, 0, stream>>>(x, xs);

  moe_gate_kernel<<<Ntok / 8, 256, 0, stream>>>(x, gw, gb, counts, sum_p, psum, p2sum, idxbuf);
  moe_ffn_kernel<<<dim3(Ee, Ntok / MT), 256, 0, stream>>>(xs, w1s, b1, w2s, b2,
                                                          counts, sum_p, idxbuf, out, ybf);
  moe_orth_kernel<<<dim3(Ee, Ntok / 16), 32, 0, stream>>>((const unsigned*)ybf, counts,
                                                          idxbuf, offac);
  moe_loss_kernel<<<1, 32, 0, stream>>>(counts, psum, p2sum, offac, loss_out);
}
